// SelfAttention_16071767622056
// MI455X (gfx1250) — matrix operand order repaired and checked by reading, compiled, not run on hardware
//
#include <hip/hip_runtime.h>
#include <hip/hip_bf16.h>

// ---------------------------------------------------------------------------
// SAGAN self-attention for MI455X (gfx1250), flash-attention formulation.
//   B=8, C=512, N=64*64=4096, d_qk=64.
//   Q = g^T (wk), K = f^T (wq), V = h^T (wv); softmax over keys i.
// GEMMs via v_wmma_f32_16x16x32_bf16; V staged to LDS with the Tensor Data
// Mover (tensor_load_to_lds, TENSORcnt), double-buffered against compute.
// ---------------------------------------------------------------------------

#define B_DIM 8
#define C_DIM 512
#define N_SEQ 4096
#define D_QK  64
#define VROW  40   // padded LDS row pitch (elements) -> conflict-free b128 reads

typedef __attribute__((ext_vector_type(8)))  __bf16 v8bf;
typedef __attribute__((ext_vector_type(16))) __bf16 v16bf;
typedef __attribute__((ext_vector_type(8)))  float  v8f;
typedef __attribute__((ext_vector_type(4)))  unsigned v4u;
typedef __attribute__((ext_vector_type(8)))  unsigned v8u;

// ---------------------------------------------------------------------------
// Fragment loaders (CDNA5 16-bit WMMA layouts, CDNA5 ISA §7.12.2)
// A (16x32 MxK): lane<16 -> row=lane, K {0..7},{16..23}; lane>=16 -> K +8.
//   => two contiguous 16B chunks per lane from row-major [M][K].
// B (32x16 KxN): lane holds column N=lane%16, K {0..7},{16..23}; lane>=16 -> K +8
//   => two contiguous 16B chunks per lane from B^T row-major [N][K].
// ---------------------------------------------------------------------------
__device__ __forceinline__ v16bf load_a_frag(const __bf16* base, int ld) {
  const int lane = threadIdx.x & 31;
  const int l = lane & 15, hi = lane >> 4;
  const __bf16* p = base + (size_t)l * ld + hi * 8;
  v8bf c0 = *(const v8bf*)(p);
  v8bf c1 = *(const v8bf*)(p + 16);
  return __builtin_shufflevector(c0, c1, 0, 1, 2, 3, 4, 5, 6, 7,
                                          8, 9, 10, 11, 12, 13, 14, 15);
}

__device__ __forceinline__ v16bf load_b_frag(const __bf16* baseT, int ld) {
  const int lane = threadIdx.x & 31;
  const int l = lane & 15, hi = lane >> 4;
  const __bf16* p = baseT + (size_t)l * ld + hi * 8;
  v8bf c0 = *(const v8bf*)(p);
  v8bf c1 = *(const v8bf*)(p + 16);
  return __builtin_shufflevector(c0, c1, 0, 1, 2, 3, 4, 5, 6, 7,
                                          8, 9, 10, 11, 12, 13, 14, 15);
}

__device__ __forceinline__ v8f wmma_bf16(v16bf a, v16bf b, v8f c) {
  return __builtin_amdgcn_wmma_f32_16x16x32_bf16(
      false, a, false, b, (short)0, c, false, false);
}

// ---------------------------------------------------------------------------
// Tensor Data Mover: DMA one V block Vt[0..511][kb..kb+32) (bf16) into LDS,
// padding each 64B row with 16B so the LDS pitch is VROW=40 elements.
// D# packing per CDNA5 ISA (2D tile, groups 0-1).
// ---------------------------------------------------------------------------
__device__ __forceinline__ void tdm_load_vblock(const __bf16* gsrc,
                                                unsigned lds_off) {
  const unsigned long long ga = (unsigned long long)(uintptr_t)gsrc;
  const v4u g0 = {
      1u,                                   // count=1, user descriptor
      lds_off,                              // lds_addr (bytes)
      (unsigned)ga,                         // global_addr[31:0]
      ((unsigned)(ga >> 32) & 0x01FFFFFFu)  // global_addr[56:32]
          | 0x80000000u                     // type=2 ("image")
  };
  const v8u g1 = {
      // wg_mask=0 | data_size=1(2B)<<16 | pad_en<<20 | pad_int=3<<22 |
      // pad_amt=3<<25  (pad 4 dwords after every 16 dwords -> pitch 80B)
      0x06D10000u,
      (N_SEQ & 0xFFFFu) << 16,              // tensor_dim0[15:0] << 16
      (N_SEQ >> 16) | ((C_DIM & 0xFFFFu) << 16),  // dim0 hi | dim1 lo
      (C_DIM >> 16) | (32u << 16),          // dim1 hi | tile_dim0=32
      (unsigned)C_DIM,                      // tile_dim1=512, tile_dim2=0
      (unsigned)N_SEQ,                      // tensor_dim0_stride[31:0]
      0u, 0u                                // stride0 hi, dim1_stride
  };
  asm volatile("tensor_load_to_lds %0, %1" :: "s"(g0), "s"(g1) : "memory");
}

// ---------------------------------------------------------------------------
// Stage 0: weight convert fp32 -> bf16 (row-major [o][c] == B^T layout)
// ---------------------------------------------------------------------------
__global__ void cvt_bf16_kernel(const float* __restrict__ w,
                                __bf16* __restrict__ wb, int n) {
  int i = blockIdx.x * blockDim.x + threadIdx.x;
  if (i < n) wb[i] = (__bf16)w[i];
}

// ---------------------------------------------------------------------------
// Stage 1: x[b][c][n] fp32 -> xT[b][n][c] bf16, LDS-tiled 32x32 transpose
// ---------------------------------------------------------------------------
__global__ __launch_bounds__(256) void transpose_bf16_kernel(
    const float* __restrict__ x, __bf16* __restrict__ xT) {
  __shared__ __bf16 tile[32][33];
  const int b = blockIdx.z;
  const int c0 = blockIdx.y * 32;
  const int n0 = blockIdx.x * 32;
  const int tx = threadIdx.x, ty = threadIdx.y;  // blockDim = (32, 8)
  const float* xb = x + (size_t)b * C_DIM * N_SEQ;
#pragma unroll
  for (int i = 0; i < 32; i += 8)
    tile[ty + i][tx] = (__bf16)xb[(size_t)(c0 + ty + i) * N_SEQ + n0 + tx];
  __syncthreads();
  __bf16* xtb = xT + (size_t)b * N_SEQ * C_DIM;
#pragma unroll
  for (int i = 0; i < 32; i += 8)
    xtb[(size_t)(n0 + ty + i) * C_DIM + c0 + tx] = tile[tx][ty + i];
}

// ---------------------------------------------------------------------------
// Stage 2: projection GEMM. out[n][o] = sum_c xT[n][c] * w[o][c]
//   One wave per 16x16 tile, k-loop over C_DIM in steps of 32.
//   transOut=0: out[b][n][O] (Q/K);  transOut=1: out[b][o][N] (Vt)
// ---------------------------------------------------------------------------
__global__ __launch_bounds__(256) void proj_gemm_kernel(
    const __bf16* __restrict__ xT, const __bf16* __restrict__ wT,
    __bf16* __restrict__ out, int O, int transOut) {
  const int b = blockIdx.z;
  const int wave = threadIdx.x >> 5;
  const int lane = threadIdx.x & 31;
  const int l = lane & 15, hi = lane >> 4;
  const int nt = blockIdx.y * 8 + wave;
  const int ot = blockIdx.x;

  const __bf16* xb = xT + (size_t)b * N_SEQ * C_DIM + (size_t)nt * 16 * C_DIM;
  const __bf16* wrow = wT + (size_t)ot * 16 * C_DIM;

  v8f acc = {};
#pragma unroll 4
  for (int k = 0; k < C_DIM; k += 32) {
    v16bf a = load_a_frag(xb + k, C_DIM);
    v16bf bf = load_b_frag(wrow + k, C_DIM);
    acc = wmma_bf16(a, bf, acc);
  }

  if (transOut) {
    const int c = ot * 16 + l;
    __bf16* dst = out + (size_t)b * O * N_SEQ + (size_t)c * N_SEQ +
                  nt * 16 + hi * 8;
    v8bf pk;
#pragma unroll
    for (int r = 0; r < 8; r++) pk[r] = (__bf16)acc[r];
    *(v8bf*)dst = pk;
  } else {
    const int d = ot * 16 + l;
    __bf16* dst = out + (size_t)b * N_SEQ * O +
                  (size_t)(nt * 16 + hi * 8) * O + d;
#pragma unroll
    for (int r = 0; r < 8; r++) dst[(size_t)r * O] = (__bf16)acc[r];
  }
}

// ---------------------------------------------------------------------------
// Stage 3: flash attention + epilogue.
//   Grid: (N/64 query blocks, B). Block: 512 threads = 16 waves,
//   wave = (j-tile 0..3, 128-channel group 0..3); 64 f32 acc VGPRs/lane.
//   V blocks DMA'd into double-buffered LDS by the TDM (wave 0 issues),
//   overlapped with S-GEMM + online softmax. Per-row softmax stats stay
//   per-lane via 16-lane shfl_xor reductions (C/D layout alignment).
// ---------------------------------------------------------------------------
__global__ __launch_bounds__(512) void flash_attn_kernel(
    const __bf16* __restrict__ Qm, const __bf16* __restrict__ Km,
    const __bf16* __restrict__ Vt, const float* __restrict__ x,
    const float* __restrict__ gamma, float* __restrict__ out) {
  __shared__ __bf16 vbuf[2][C_DIM * VROW];   // 2 x 40KB V tiles (padded pitch)
  __shared__ __bf16 plds[16 * 16 * VROW];    // per-wave P transpose slabs

  const int b = blockIdx.y;
  const int wave = threadIdx.x >> 5;
  const int lane = threadIdx.x & 31;
  const int l = lane & 15, hi = lane >> 4;
  const int jt = wave & 3;
  const int cg = wave >> 2;
  const int j0 = blockIdx.x * 64 + jt * 16;

  const __bf16* Qb = Qm + (size_t)b * N_SEQ * D_QK;
  const __bf16* Kb = Km + (size_t)b * N_SEQ * D_QK;
  const __bf16* Vb = Vt + (size_t)b * C_DIM * N_SEQ;

  const unsigned lds_v0 = (unsigned)(uintptr_t)(&vbuf[0][0]);
  const unsigned lds_v1 = (unsigned)(uintptr_t)(&vbuf[1][0]);

  // Prologue: DMA first V block into buffer 0 (one TDM op per workgroup).
  if (wave == 0) tdm_load_vblock(Vb, lds_v0);

  const v16bf qa0 = load_a_frag(Qb + (size_t)j0 * D_QK + 0, D_QK);
  const v16bf qa1 = load_a_frag(Qb + (size_t)j0 * D_QK + 32, D_QK);

  v8f acc[8];
#pragma unroll
  for (int t = 0; t < 8; t++) acc[t] = (v8f){};
  float mrow[8], lrow[8];
#pragma unroll
  for (int r = 0; r < 8; r++) { mrow[r] = -1e30f; lrow[r] = 0.0f; }

  __bf16* myP = plds + wave * (16 * VROW);

  for (int kb = 0; kb < N_SEQ; kb += 32) {
    const int cur = (kb >> 5) & 1;
    const bool have_next = (kb + 32) < N_SEQ;
    // Kick off DMA of the next V block into the other buffer (its readers
    // finished at the trailing barrier of the previous iteration).
    if (wave == 0 && have_next)
      tdm_load_vblock(Vb + kb + 32, cur ? lds_v0 : lds_v1);
    if (have_next)
      __builtin_prefetch(Kb + (size_t)(kb + 32) * D_QK, 0, 1);

    // ---- S = Q K^T for two 16x16 i-tiles (contraction = 64) ----
    v8f s[2];
#pragma unroll
    for (int it = 0; it < 2; ++it) {
      const __bf16* krow = Kb + (size_t)(kb + it * 16) * D_QK;
      v16bf kb0 = load_b_frag(krow + 0, D_QK);
      v16bf kb1 = load_b_frag(krow + 32, D_QK);
      v8f z = {};
      z = wmma_bf16(qa0, kb0, z);
      z = wmma_bf16(qa1, kb1, z);
      s[it] = z;
    }

    // ---- online softmax over this block's 32 keys ----
    float newm[8];
#pragma unroll
    for (int r = 0; r < 8; r++) {
      float v = fmaxf(s[0][r], s[1][r]);
      v = fmaxf(v, __shfl_xor(v, 1));
      v = fmaxf(v, __shfl_xor(v, 2));
      v = fmaxf(v, __shfl_xor(v, 4));
      v = fmaxf(v, __shfl_xor(v, 8));
      newm[r] = fmaxf(mrow[r], v);
    }
#pragma unroll
    for (int r = 0; r < 8; r++) {
      float sc = __expf(mrow[r] - newm[r]);
      mrow[r] = newm[r];
      lrow[r] *= sc;
#pragma unroll
      for (int t = 0; t < 8; t++) acc[t][r] *= sc;
    }

    // ---- P = exp(S-m); row sums; D-layout -> A-layout via private LDS ----
#pragma unroll
    for (int r = 0; r < 8; r++) {
      float p0 = __expf(s[0][r] - mrow[r]);
      float p1 = __expf(s[1][r] - mrow[r]);
      float ps = p0 + p1;
      ps += __shfl_xor(ps, 1);
      ps += __shfl_xor(ps, 2);
      ps += __shfl_xor(ps, 4);
      ps += __shfl_xor(ps, 8);
      lrow[r] += ps;
      const int row = hi * 8 + r;
      myP[row * VROW + l] = (__bf16)p0;
      myP[row * VROW + 16 + l] = (__bf16)p1;
    }
    v16bf pa = load_a_frag(myP, VROW);  // same-wave LDS ops are in order

    // ---- current V block must be resident: wave0 allows 1 outstanding
    //      (the prefetched next block), others just rendezvous ----
    if (wave == 0) {
      if (have_next) __builtin_amdgcn_s_wait_tensorcnt(1);
      else           __builtin_amdgcn_s_wait_tensorcnt(0);
    }
    __syncthreads();

    // ---- O += P * V from LDS (8 channel tiles per wave) ----
    const __bf16* ldsV = &vbuf[cur][0];
#pragma unroll
    for (int t = 0; t < 8; t++) {
      const int c = cg * 128 + t * 16;
      v16bf vb = load_b_frag(ldsV + (size_t)c * VROW, VROW);
      acc[t] = wmma_bf16(pa, vb, acc[t]);
    }
    __syncthreads();  // all waves done with vbuf[cur] before it is refilled
  }

  // ---- epilogue: out[b][c][j] = gamma * (acc/l) + x[b][c][j] ----
  const float g = gamma[0];
  const float* xb = x + (size_t)b * C_DIM * N_SEQ;
  float* ob = out + (size_t)b * C_DIM * N_SEQ;
  const int jrow = j0 + hi * 8;
#pragma unroll
  for (int t = 0; t < 8; t++) {
    const int c = cg * 128 + t * 16 + l;
    const float* xr = xb + (size_t)c * N_SEQ + jrow;
    float* orow = ob + (size_t)c * N_SEQ + jrow;
#pragma unroll
    for (int r = 0; r < 8; r++)
      orow[r] = g * (acc[t][r] / lrow[r]) + xr[r];
  }
}

// ---------------------------------------------------------------------------
// Host launcher
// ---------------------------------------------------------------------------
extern "C" void kernel_launch(void* const* d_in, const int* in_sizes, int n_in,
                              void* d_out, int out_size, void* d_ws,
                              size_t ws_size, hipStream_t stream) {
  (void)in_sizes; (void)n_in; (void)out_size; (void)ws_size;
  const float* x = (const float*)d_in[0];
  const float* wq = (const float*)d_in[1];   // -> f (keys i)
  const float* wk = (const float*)d_in[2];   // -> g (queries j)
  const float* wv = (const float*)d_in[3];   // -> h (values)
  const float* gamma = (const float*)d_in[4];
  float* out = (float*)d_out;

  char* ws = (char*)d_ws;
  size_t off = 0;
  auto carve = [&](size_t bytes) -> void* {
    void* p = ws + off;
    off += (bytes + 255) & ~(size_t)255;
    return p;
  };
  __bf16* xT  = (__bf16*)carve((size_t)B_DIM * N_SEQ * C_DIM * 2);  // 32 MB
  __bf16* Qm  = (__bf16*)carve((size_t)B_DIM * N_SEQ * D_QK * 2);   //  4 MB
  __bf16* Km  = (__bf16*)carve((size_t)B_DIM * N_SEQ * D_QK * 2);   //  4 MB
  __bf16* Vtm = (__bf16*)carve((size_t)B_DIM * C_DIM * N_SEQ * 2);  // 32 MB
  __bf16* wqb = (__bf16*)carve((size_t)D_QK * C_DIM * 2);
  __bf16* wkb = (__bf16*)carve((size_t)D_QK * C_DIM * 2);
  __bf16* wvb = (__bf16*)carve((size_t)C_DIM * C_DIM * 2);

  cvt_bf16_kernel<<<(D_QK * C_DIM + 255) / 256, 256, 0, stream>>>(wq, wqb, D_QK * C_DIM);
  cvt_bf16_kernel<<<(D_QK * C_DIM + 255) / 256, 256, 0, stream>>>(wk, wkb, D_QK * C_DIM);
  cvt_bf16_kernel<<<(C_DIM * C_DIM + 255) / 256, 256, 0, stream>>>(wv, wvb, C_DIM * C_DIM);

  transpose_bf16_kernel<<<dim3(N_SEQ / 32, C_DIM / 32, B_DIM), dim3(32, 8), 0,
                          stream>>>(x, xT);

  proj_gemm_kernel<<<dim3(D_QK / 16, N_SEQ / 16 / 8, B_DIM), 256, 0, stream>>>(
      xT, wqb, Km, D_QK, 0);
  proj_gemm_kernel<<<dim3(D_QK / 16, N_SEQ / 16 / 8, B_DIM), 256, 0, stream>>>(
      xT, wkb, Qm, D_QK, 0);
  proj_gemm_kernel<<<dim3(C_DIM / 16, N_SEQ / 16 / 8, B_DIM), 256, 0, stream>>>(
      xT, wvb, Vtm, C_DIM, 1);

  flash_attn_kernel<<<dim3(N_SEQ / 64, B_DIM), 512, 0, stream>>>(
      Qm, Km, Vtm, x, gamma, out);
}
